// PyMGRNN_30090540875803
// MI455X (gfx1250) — matrix operand order repaired and checked by reading, not compiled, not run on hardware
//
#include <hip/hip_runtime.h>
#include <hip/hip_bf16.h>
#include <math.h>

// Problem constants (match reference)
#define V_SIZE 50257
#define V_PAD  50432   // next multiple of 256 (GEMM N-tile)
#define D_EMB  512
#define H_DIM  1024
#define B_SZ   8
#define T_LEN  512
#define ROWS   (B_SZ * T_LEN)   // 4096

typedef __bf16 bf16_t;
typedef __attribute__((ext_vector_type(16))) __bf16 bf16x16;
typedef __attribute__((ext_vector_type(8)))  __bf16 bf16x8;
typedef __attribute__((ext_vector_type(8)))  float  f32x8;

// ---------------------------------------------------------------------------
// gfx1250 async global->LDS copy (ASYNCcnt-tracked DMA), per-lane 16 bytes.
// VDST VGPR holds the LDS byte address (wave-relative), VADDR the 64-bit
// global address.  Completion enforced with s_wait_asynccnt before the
// workgroup barrier.
// ---------------------------------------------------------------------------
__device__ __forceinline__ void async_copy_b128(const void* g, unsigned lds_byte) {
  asm volatile("global_load_async_to_lds_b128 %0, %1, off"
               :: "v"(lds_byte), "v"((unsigned long long)(uintptr_t)g)
               : "memory");
}
__device__ __forceinline__ void wait_async0() {
  asm volatile("s_wait_asynccnt 0x0" ::: "memory");
}

// ---------------------------------------------------------------------------
// WMMA fragment helpers.
// A-layout (16-bit 16x32, as loaded here): lane l (m = l%16, c = l/16) holds
//   K = [k0 + c*16 .. +15] of row m (contiguous, same k order as B).
// B-layout (32x16): lane l (n = l%16, c = l/16) holds K = [k0 + c*16 .. +15]
//   of column n  ==> contiguous 32B from B^T stored as [N][K] row-major.
// ---------------------------------------------------------------------------
__device__ __forceinline__ bf16x16 combine16(bf16x8 lo, bf16x8 hi) {
  return __builtin_shufflevector(lo, hi, 0, 1, 2, 3, 4, 5, 6, 7,
                                 8, 9, 10, 11, 12, 13, 14, 15);
}

__device__ __forceinline__ bf16x16 load_b_frag(const bf16_t* __restrict__ Bt,
                                               int ldb, int col_base, int k0,
                                               int lane) {
  int n = lane & 15, c = lane >> 4;
  const bf16_t* p = Bt + (size_t)(col_base + n) * ldb + k0 + c * 16;
  return *(const bf16x16*)p;
}

__device__ __forceinline__ f32x8 wmma_bf16(bf16x16 a, bf16x16 b, f32x8 c) {
  return __builtin_amdgcn_wmma_f32_16x16x32_bf16(false, a, false, b,
                                                 (short)0, c, false, false);
}

// ---------------------------------------------------------------------------
// f32 [K][N] -> bf16 [Npad][K]  (transpose + convert, zero-fill pad rows)
// ---------------------------------------------------------------------------
__global__ __launch_bounds__(256) void transpose_to_bf16(
    const float* __restrict__ src, bf16_t* __restrict__ dst,
    int K, int N, int Npad) {
  __shared__ float tile[32][33];
  int n0 = blockIdx.x * 32, k0 = blockIdx.y * 32;
  int tx = threadIdx.x, ty = threadIdx.y;  // block (32, 8)
#pragma unroll
  for (int i = 0; i < 4; ++i) {
    int k = k0 + ty + i * 8, n = n0 + tx;
    tile[ty + i * 8][tx] = (n < N) ? src[(size_t)k * N + n] : 0.0f;
  }
  __syncthreads();
#pragma unroll
  for (int i = 0; i < 4; ++i) {
    int n = n0 + ty + i * 8, k = k0 + tx;
    dst[(size_t)n * K + k] = (bf16_t)tile[tx][ty + i * 8];
  }
}

// ---------------------------------------------------------------------------
// Embedding gather: xe[t*B + b][d] = bf16(E[x[b][t]][d])  (time-major rows)
// ---------------------------------------------------------------------------
__global__ __launch_bounds__(256) void embed_gather(
    const int* __restrict__ x, const float* __restrict__ E,
    bf16_t* __restrict__ xe) {
  size_t idx = (size_t)blockIdx.x * blockDim.x + threadIdx.x;
  if (idx >= (size_t)ROWS * D_EMB) return;
  int d = (int)(idx % D_EMB);
  int row = (int)(idx / D_EMB);  // row = t*B + b
  int t = row / B_SZ, b = row % B_SZ;
  int tok = x[b * T_LEN + t];
  xe[idx] = (bf16_t)E[(size_t)tok * D_EMB + d];
}

// ---------------------------------------------------------------------------
// C[M][ldc](f32) = A[M][K](bf16) @ Bt[Npad][K](bf16)^T (+bias).
// WG = 256 thr (8 waves, 2Mx4N), WG tile 128x256, per-wave 64x64 (4x4 frags,
// 16 f32x8 accumulators).  A K-slab (128x32) is staged into LDS with
// double-buffered global_load_async_to_lds_b128 so the DMA overlaps WMMA;
// B fragments are register-resident direct global loads with prefetch.
// ---------------------------------------------------------------------------
#define APAD 40  // padded row length (elements) to spread LDS banks
__global__ __launch_bounds__(256) void gemm_bf16_wmma(
    const bf16_t* __restrict__ A, const bf16_t* __restrict__ Bt,
    float* __restrict__ C, const float* __restrict__ bias,
    int M, int Nreal, int K, int ldc) {
  __shared__ __align__(16) bf16_t atile[2][128][APAD];  // 20 KB

  int lane = threadIdx.x & 31, wave = threadIdx.x >> 5;
  int wave_m = wave & 1, wave_n = wave >> 1;
  int mblock = blockIdx.y * 128;
  int mtile = wave_m * 64;                          // LDS-row base for wave
  int ntile = blockIdx.x * 256 + wave_n * 64;       // global cols [ntile,+64)
  int m16 = lane & 15, c = lane >> 4;

  f32x8 acc[4][4];
#pragma unroll
  for (int i = 0; i < 4; ++i)
#pragma unroll
    for (int j = 0; j < 4; ++j) acc[i][j] = (f32x8)0.0f;

  // Stage A[mblock..+128][k0..+32) into LDS buffer `buf` (async DMA).
  auto stage_a = [&](int buf, int k0) {
    int c0 = (int)threadIdx.x * 2;
#pragma unroll
    for (int q = 0; q < 2; ++q) {
      int cc = c0 + q;              // 512 chunks of 16B
      int row = cc >> 2, kb = (cc & 3) * 16;
      const bf16_t* gp = A + (size_t)(mblock + row) * K + k0;
      unsigned l = (unsigned)(uintptr_t)&atile[buf][row][0] + (unsigned)kb;
      async_copy_b128((const char*)gp + kb, l);
    }
  };

  int nk = K / 32;
  stage_a(0, 0);
  wait_async0();
  __syncthreads();

  int buf = 0;
  for (int kt = 0; kt < nk; ++kt) {
    int k0 = kt * 32;
    if (kt + 1 < nk) {
      stage_a(buf ^ 1, k0 + 32);  // overlap DMA with WMMA below
      __builtin_prefetch(Bt + (size_t)(ntile + m16) * K + k0 + 32, 0, 1);
    }
    bf16x16 af[4];
#pragma unroll
    for (int i = 0; i < 4; ++i) {
      const bf16_t* p = &atile[buf][mtile + i * 16 + m16][c * 16];
      af[i] = combine16(*(const bf16x8*)p, *(const bf16x8*)(p + 8));
    }
#pragma unroll
    for (int j = 0; j < 4; ++j) {
      bf16x16 b = load_b_frag(Bt, K, ntile + j * 16, k0, lane);
#pragma unroll
      for (int i = 0; i < 4; ++i) acc[i][j] = wmma_bf16(af[i], b, acc[i][j]);
    }
    wait_async0();
    __syncthreads();
    buf ^= 1;
  }

  int mo = 8 * c;
#pragma unroll
  for (int j = 0; j < 4; ++j) {
    int col = ntile + j * 16 + m16;
    if (col < Nreal) {
      float bv = bias ? bias[col] : 0.0f;
#pragma unroll
      for (int i = 0; i < 4; ++i) {
#pragma unroll
        for (int r = 0; r < 8; ++r) {
          int row = mblock + mtile + i * 16 + mo + r;
          if (row < M) C[(size_t)row * ldc + col] = acc[i][j][r] + bv;
        }
      }
    }
  }
}

// ---------------------------------------------------------------------------
// Sequential RNN scan for one layer:  h_{t} = tanh(pre[t] + h_{t-1} @ Whh)
// One workgroup, 512 threads = 16 waves; h (16x1024, rows>=B zero-padded)
// double-buffered in LDS. Each wave owns 64 columns (4 N-fragments).
// yout[b*stride_b + t*stride_t + col] = h_t[b][col] as bf16.
// ---------------------------------------------------------------------------
__global__ __launch_bounds__(512) void rnn_scan_wmma(
    const float* __restrict__ pre, const bf16_t* __restrict__ Whh_t,
    bf16_t* __restrict__ yout, long stride_b, long stride_t) {
  __shared__ __align__(32) bf16_t hbuf[2][16][H_DIM];  // 64 KB

  for (int i = threadIdx.x; i < 2 * 16 * H_DIM; i += 512)
    ((bf16_t*)hbuf)[i] = (bf16_t)0.0f;
  __syncthreads();

  int lane = threadIdx.x & 31, wave = threadIdx.x >> 5;
  int nbase = wave * 64;
  int m16 = lane & 15, c = lane >> 4;

  for (int t = 0; t < T_LEN; ++t) {
    int cur = t & 1, nxt = cur ^ 1;
    f32x8 acc[4];
#pragma unroll
    for (int j = 0; j < 4; ++j) acc[j] = (f32x8)0.0f;

    for (int kt = 0; kt < H_DIM / 32; ++kt) {
      int k0 = kt * 32;
      const bf16_t* hp = &hbuf[cur][m16][k0 + c * 16];
      bf16x16 a = combine16(*(const bf16x8*)hp, *(const bf16x8*)(hp + 8));
#pragma unroll
      for (int j = 0; j < 4; ++j) {
        bf16x16 b = load_b_frag(Whh_t, H_DIM, nbase + j * 16, k0, lane);
        acc[j] = wmma_bf16(a, b, acc[j]);
      }
    }

#pragma unroll
    for (int j = 0; j < 4; ++j) {
      int col = nbase + j * 16 + m16;
#pragma unroll
      for (int r = 0; r < 8; ++r) {
        int m = r + 8 * c;
        if (m < B_SZ) {
          float v = tanhf(pre[((size_t)t * B_SZ + m) * H_DIM + col] + acc[j][r]);
          bf16_t bv = (bf16_t)v;
          hbuf[nxt][m][col] = bv;
          yout[(size_t)m * stride_b + (size_t)t * stride_t + col] = bv;
        }
      }
    }
    __syncthreads();
  }
}

// ---------------------------------------------------------------------------
// Cross-entropy: per-row logsumexp + nll, then reduction to scalar loss.
// ---------------------------------------------------------------------------
__global__ __launch_bounds__(256) void ce_row(
    const float* __restrict__ logits, const int* __restrict__ y,
    float* __restrict__ nll) {
  __shared__ float red[256];
  int r = blockIdx.x, tid = threadIdx.x;
  const float* row = logits + (size_t)r * V_SIZE;

  float mx = -INFINITY;
  for (int i = tid; i < V_SIZE; i += 256) mx = fmaxf(mx, row[i]);
  red[tid] = mx; __syncthreads();
  for (int s = 128; s > 0; s >>= 1) {
    if (tid < s) red[tid] = fmaxf(red[tid], red[tid + s]);
    __syncthreads();
  }
  float mall = red[0]; __syncthreads();

  float sum = 0.0f;
  for (int i = tid; i < V_SIZE; i += 256) sum += __expf(row[i] - mall);
  red[tid] = sum; __syncthreads();
  for (int s = 128; s > 0; s >>= 1) {
    if (tid < s) red[tid] += red[tid + s];
    __syncthreads();
  }
  if (tid == 0) {
    int yy = y[r];  // row r == flat (b*T + t) index of y
    nll[r] = (yy != -1) ? -(row[yy] - mall - __logf(red[0])) : 0.0f;
  }
}

__global__ __launch_bounds__(256) void ce_reduce(
    const float* __restrict__ nll, const int* __restrict__ y,
    float* __restrict__ loss_out) {
  __shared__ float rs[256];
  __shared__ int rc[256];
  int tid = threadIdx.x;
  float s = 0.0f; int cnt = 0;
  for (int i = tid; i < ROWS; i += 256) {
    s += nll[i];
    cnt += (y[i] != -1) ? 1 : 0;
  }
  rs[tid] = s; rc[tid] = cnt; __syncthreads();
  for (int st = 128; st > 0; st >>= 1) {
    if (tid < st) { rs[tid] += rs[tid + st]; rc[tid] += rc[tid + st]; }
    __syncthreads();
  }
  if (tid == 0) loss_out[0] = rs[0] / (float)(rc[0] > 0 ? rc[0] : 1);
}

// ---------------------------------------------------------------------------
extern "C" void kernel_launch(void* const* d_in, const int* in_sizes, int n_in,
                              void* d_out, int out_size, void* d_ws,
                              size_t ws_size, hipStream_t stream) {
  const int*   x    = (const int*)d_in[0];
  const int*   y    = (const int*)d_in[1];
  const float* E    = (const float*)d_in[2];
  const float* Wih0 = (const float*)d_in[3];
  const float* Whh0 = (const float*)d_in[4];
  const float* Wih1 = (const float*)d_in[5];
  const float* Whh1 = (const float*)d_in[6];
  const float* Wfc  = (const float*)d_in[7];
  const float* bfc  = (const float*)d_in[8];
  float* logits = (float*)d_out;
  float* loss   = (float*)d_out + (size_t)ROWS * V_SIZE;

  // Workspace layout (256-byte aligned slices)
  char* ws = (char*)d_ws;
  size_t off = 0;
  auto alloc = [&](size_t bytes) -> char* {
    char* p = ws + off;
    off = (off + bytes + 255) & ~(size_t)255;
    return p;
  };
  bf16_t* Wih0t = (bf16_t*)alloc((size_t)H_DIM * D_EMB * 2);
  bf16_t* Whh0t = (bf16_t*)alloc((size_t)H_DIM * H_DIM * 2);
  bf16_t* Wih1t = (bf16_t*)alloc((size_t)H_DIM * H_DIM * 2);
  bf16_t* Whh1t = (bf16_t*)alloc((size_t)H_DIM * H_DIM * 2);
  bf16_t* Wfct  = (bf16_t*)alloc((size_t)V_PAD * H_DIM * 2);
  bf16_t* xe    = (bf16_t*)alloc((size_t)ROWS * D_EMB * 2);
  float*  pre0  = (float*) alloc((size_t)ROWS * H_DIM * 4);
  bf16_t* y0    = (bf16_t*)alloc((size_t)ROWS * H_DIM * 2);
  float*  pre1  = (float*) alloc((size_t)ROWS * H_DIM * 4);
  bf16_t* feats = (bf16_t*)alloc((size_t)ROWS * H_DIM * 2);
  float*  nll   = (float*) alloc((size_t)ROWS * 4);

  dim3 tb(32, 8);
  // Weight transpose+convert: src f32 [K][N] -> dst bf16 [Npad][K]
  transpose_to_bf16<<<dim3(H_DIM / 32, D_EMB / 32), tb, 0, stream>>>(
      Wih0, Wih0t, D_EMB, H_DIM, H_DIM);
  transpose_to_bf16<<<dim3(H_DIM / 32, H_DIM / 32), tb, 0, stream>>>(
      Whh0, Whh0t, H_DIM, H_DIM, H_DIM);
  transpose_to_bf16<<<dim3(H_DIM / 32, H_DIM / 32), tb, 0, stream>>>(
      Wih1, Wih1t, H_DIM, H_DIM, H_DIM);
  transpose_to_bf16<<<dim3(H_DIM / 32, H_DIM / 32), tb, 0, stream>>>(
      Whh1, Whh1t, H_DIM, H_DIM, H_DIM);
  transpose_to_bf16<<<dim3(V_PAD / 32, H_DIM / 32), tb, 0, stream>>>(
      Wfc, Wfct, H_DIM, V_SIZE, V_PAD);

  embed_gather<<<(ROWS * D_EMB + 255) / 256, 256, 0, stream>>>(x, E, xe);

  // pre0 = xe @ Wih0   (time-major rows)
  gemm_bf16_wmma<<<dim3(H_DIM / 256, ROWS / 128), 256, 0, stream>>>(
      xe, Wih0t, pre0, nullptr, ROWS, H_DIM, D_EMB, H_DIM);
  // layer-0 scan: y0 time-major (row = t*B + b)
  rnn_scan_wmma<<<1, 512, 0, stream>>>(pre0, Whh0t, y0,
                                       (long)H_DIM, (long)B_SZ * H_DIM);
  // pre1 = y0 @ Wih1
  gemm_bf16_wmma<<<dim3(H_DIM / 256, ROWS / 128), 256, 0, stream>>>(
      y0, Wih1t, pre1, nullptr, ROWS, H_DIM, H_DIM, H_DIM);
  // layer-1 scan: feats batch-major (row = b*T + t) to match logits layout
  rnn_scan_wmma<<<1, 512, 0, stream>>>(pre1, Whh1t, feats,
                                       (long)T_LEN * H_DIM, (long)H_DIM);
  // logits = feats @ Wfc + bfc   (dominant GEMM, 422 GFLOP)
  gemm_bf16_wmma<<<dim3(V_PAD / 256, ROWS / 128), 256, 0, stream>>>(
      feats, Wfct, logits, bfc, ROWS, V_SIZE, H_DIM, V_SIZE);

  ce_row<<<ROWS, 256, 0, stream>>>(logits, y, nll);
  ce_reduce<<<1, 256, 0, stream>>>(nll, y, loss);
  (void)in_sizes; (void)n_in; (void)out_size; (void)ws_size;
}
